// LinearKANExtended_15350213116477
// MI455X (gfx1250) — matrix operand order repaired and checked by reading, not compiled, not run on hardware
//
#include <hip/hip_runtime.h>
#include <hip/hip_bf16.h>

// ---------------------------------------------------------------------------
// LinearKAN-extended:  out[b,o] = sum_g w[b,g] * (x[b,:] . coeff[g,o,:]) + bias[o]
// B = 16384, I = O = 1024, G = 9.
// Compute-bound -> bf16 WMMA (16x16x32) with fp32 accumulation.
// Optimizations this round:
//   * 32x64 wave tile (m=2, n=4): 1.5 b128 loads per WMMA (was 2.5)
//   * 4 waves stacked in M share one 64-col B stream (B hits L1)
//   * per-block segment skipping: hat weights have support +-0.25 around
//     x_mean ~ N(0, 0.031), so ~6 of 9 segment GEMMs are exactly zero-weight
//     for every row in a block and are skipped (uniform branch, EXEC all-1s)
// ---------------------------------------------------------------------------

#define B_ROWS   16384
#define IN_DIM   1024
#define OUT_DIM  1024
#define NSEG     9           // num_segments + 1 grid points

#define BLK_M    128         // rows per block (4 waves x 32)
#define BLK_N    64          // cols per block
#define WAVE_M   32          // rows per wave (2 subtiles of 16)

typedef __attribute__((ext_vector_type(16))) __bf16 v16bf;
typedef __attribute__((ext_vector_type(8)))  __bf16 v8bf;
typedef __attribute__((ext_vector_type(8)))  float  v8f;

// ---------------------------------------------------------------------------
// Kernel 1: per-row mean -> normalized triangular hat weights (fp32, exact),
// and convert the x row to bf16.  One block (256 threads) per row.
// ---------------------------------------------------------------------------
__global__ __launch_bounds__(256) void prep_x_kernel(const float* __restrict__ x,
                                                     __hip_bfloat16* __restrict__ xb,
                                                     float* __restrict__ wg) {
    const int row = blockIdx.x;
    const int tid = threadIdx.x;
    const float* xr = x + (size_t)row * IN_DIM;
    __hip_bfloat16* xbr = xb + (size_t)row * IN_DIM;

    float s = 0.0f;
    #pragma unroll
    for (int i = tid; i < IN_DIM; i += 256) {
        float v = xr[i];
        s += v;
        xbr[i] = __float2bfloat16(v);
    }

    __shared__ float red[256];
    red[tid] = s;
    __syncthreads();
    #pragma unroll
    for (int st = 128; st > 0; st >>= 1) {
        if (tid < st) red[tid] += red[tid + st];
        __syncthreads();
    }

    if (tid == 0) {
        const float mean = red[0] * (1.0f / (float)IN_DIM);
        const float h = 0.25f;  // (1 - (-1)) / 8
        float w[NSEG];
        float tot = 0.0f;
        #pragma unroll
        for (int g = 0; g < NSEG; ++g) {
            float grid = -1.0f + h * (float)g;
            float d = fabsf(mean - grid);
            float ww = fmaxf(0.0f, 1.0f - d / h);
            w[g] = ww;
            tot += ww;
        }
        const float inv = 1.0f / (tot + 1e-8f);
        #pragma unroll
        for (int g = 0; g < NSEG; ++g)
            wg[(size_t)row * NSEG + g] = w[g] * inv;
    }
}

// ---------------------------------------------------------------------------
// Kernel 2: coeff fp32 -> bf16 (9*1024*1024 elements; 4 per thread)
// ---------------------------------------------------------------------------
__global__ __launch_bounds__(256) void conv_coeff_kernel(const float* __restrict__ c,
                                                         __hip_bfloat16* __restrict__ cb) {
    const size_t base = ((size_t)blockIdx.x * 256 + threadIdx.x) * 4;
    #pragma unroll
    for (int i = 0; i < 4; ++i)
        cb[base + i] = __float2bfloat16(c[base + i]);
}

// ---------------------------------------------------------------------------
// Kernel 3: WMMA GEMM with per-segment weighted accumulation.
//
// Block: 128 threads = 4 waves, each wave owns 32 rows; all waves share the
// same 64 output columns (B stream shared through L1).
// Wave tile: 32x64 = 2 M-subtiles x 4 N-subtiles of 16x16.
//
// Per active segment g: fp32 temp accumulators t over the K=1024 loop
// (2*4*32 = 256 WMMAs), then acc += w[g][row] * t  (w from LDS).
//
// Operand layouts (ISA 7.12.2, 16-bit, wave32):
//   A 16x32:  lane l -> row M=l%16; lanes 0-15 K={0..7,16..23},
//             lanes 16-31 K={8..15,24..31}  -> two 16B loads at +hi*8 elems.
//   B 32x16:  lane l -> col N=l%16; same K order as A
//             -> two 16B loads at +hi*8 elems.
//   C/D 16x16: lane -> N=l%16, VGPR j -> M = j + 8*(l>=16).
// ---------------------------------------------------------------------------
__global__ __launch_bounds__(128) void kan_wmma_gemm_kernel(
        const __hip_bfloat16* __restrict__ xb,
        const __hip_bfloat16* __restrict__ cb,
        const float* __restrict__ wg,
        const float* __restrict__ bias,
        float* __restrict__ out) {
    const int lane = threadIdx.x & 31;
    const int wave = threadIdx.x >> 5;
    const int lo16 = lane & 15;
    const int hi   = lane >> 4;           // 0 for lanes 0-15, 1 for 16-31

    const int row0 = blockIdx.x * BLK_M;          // block row base
    const int wrow = row0 + wave * WAVE_M;        // wave row base
    const int col0 = blockIdx.y * BLK_N;          // block col base

    // Hat weights for this block's 128 rows: wlds[g][r], plus per-segment
    // activity flags (uniform across block).
    __shared__ float wlds[NSEG][BLK_M];
    __shared__ int   gact[NSEG];
    {
        const int r = threadIdx.x;                // 128 threads -> 128 rows
        #pragma unroll
        for (int g = 0; g < NSEG; ++g)
            wlds[g][r] = wg[(size_t)(row0 + r) * NSEG + g];
    }
    __syncthreads();
    if (threadIdx.x < NSEG) {
        int any = 0;
        for (int r = 0; r < BLK_M; ++r)
            any |= (wlds[threadIdx.x][r] != 0.0f) ? 1 : 0;
        gact[threadIdx.x] = any;
    }
    __syncthreads();

    v8f acc[2][4] = {};   // fp32 output accumulators (32x64 tile)

    const __hip_bfloat16* arow0 = xb + (size_t)(wrow + lo16) * IN_DIM;        // m = 0
    const __hip_bfloat16* arow1 = xb + (size_t)(wrow + 16 + lo16) * IN_DIM;   // m = 1

    for (int g = 0; g < NSEG; ++g) {
        if (!gact[g]) continue;           // segment weight is 0 for all rows

        v8f t[2][4] = {};
        const __hip_bfloat16* cg = cb + (size_t)g * OUT_DIM * IN_DIM;

        for (int k0 = 0; k0 < IN_DIM; k0 += 32) {
            union { v16bf v; v8bf h[2]; } A0, A1;
            const __hip_bfloat16* ap0 = arow0 + k0 + hi * 8;
            const __hip_bfloat16* ap1 = arow1 + k0 + hi * 8;
            A0.h[0] = *(const v8bf*)(ap0);
            A0.h[1] = *(const v8bf*)(ap0 + 16);
            A1.h[0] = *(const v8bf*)(ap1);
            A1.h[1] = *(const v8bf*)(ap1 + 16);

            #pragma unroll
            for (int n = 0; n < 4; ++n) {
                const __hip_bfloat16* bp =
                    cg + (size_t)(col0 + n * 16 + lo16) * IN_DIM + k0 + hi * 8;
                union { v16bf v; v8bf h[2]; } Bm;
                Bm.h[0] = *(const v8bf*)(bp);
                Bm.h[1] = *(const v8bf*)(bp + 16);
                t[0][n] = __builtin_amdgcn_wmma_f32_16x16x32_bf16(
                    false, A0.v, false, Bm.v, (short)0, t[0][n], false, false);
                t[1][n] = __builtin_amdgcn_wmma_f32_16x16x32_bf16(
                    false, A1.v, false, Bm.v, (short)0, t[1][n], false, false);
            }
        }

        // acc += w[g][row] * t   (row = wave*32 + m*16 + j + 8*hi)
        #pragma unroll
        for (int m = 0; m < 2; ++m) {
            const float* wr = &wlds[g][wave * WAVE_M + m * 16 + hi * 8];
            #pragma unroll
            for (int j = 0; j < 8; ++j) {
                const float wj = wr[j];
                #pragma unroll
                for (int n = 0; n < 4; ++n)
                    acc[m][n][j] = __builtin_fmaf(wj, t[m][n][j], acc[m][n][j]);
            }
        }
    }

    // Epilogue: add bias, store fp32 result.
    #pragma unroll
    for (int n = 0; n < 4; ++n) {
        const int col = col0 + n * 16 + lo16;
        const float bz = bias[col];
        #pragma unroll
        for (int m = 0; m < 2; ++m) {
            #pragma unroll
            for (int j = 0; j < 8; ++j) {
                const int row = wrow + m * 16 + j + hi * 8;
                out[(size_t)row * OUT_DIM + col] = acc[m][n][j] + bz;
            }
        }
    }
}

// ---------------------------------------------------------------------------
// Launcher
// ---------------------------------------------------------------------------
extern "C" void kernel_launch(void* const* d_in, const int* in_sizes, int n_in,
                              void* d_out, int out_size, void* d_ws, size_t ws_size,
                              hipStream_t stream) {
    const float* x     = (const float*)d_in[0];   // (8, 2048, 1024)
    const float* coeff = (const float*)d_in[1];   // (9, 1024, 1024)
    const float* bias  = (const float*)d_in[2];   // (1024,)
    float* out = (float*)d_out;                   // (8, 2048, 1024)

    // Workspace layout (16B-aligned):
    //   xb : 16384*1024 bf16 = 32 MiB
    //   cb : 9*1024*1024 bf16 = 18 MiB
    //   wg : 16384*9 fp32     = ~0.56 MiB
    char* ws = (char*)d_ws;
    const size_t XB_BYTES = (size_t)B_ROWS * IN_DIM * sizeof(__hip_bfloat16);
    const size_t CB_BYTES = (size_t)NSEG * OUT_DIM * IN_DIM * sizeof(__hip_bfloat16);
    __hip_bfloat16* xb = (__hip_bfloat16*)ws;
    __hip_bfloat16* cb = (__hip_bfloat16*)(ws + XB_BYTES);
    float*          wg = (float*)(ws + XB_BYTES + CB_BYTES);

    prep_x_kernel<<<B_ROWS, 256, 0, stream>>>(x, xb, wg);
    conv_coeff_kernel<<<(NSEG * OUT_DIM * IN_DIM) / (256 * 4), 256, 0, stream>>>(coeff, cb);

    dim3 grid(B_ROWS / BLK_M, OUT_DIM / BLK_N);   // (128, 16)
    kan_wmma_gemm_kernel<<<grid, 128, 0, stream>>>(xb, cb, wg, bias, out);
}
